// Model_73306501808430
// MI455X (gfx1250) — matrix operand order repaired and checked by reading, not compiled, not run on hardware
//
#include <hip/hip_runtime.h>
#include <hip/hip_bf16.h>

typedef __bf16 bf16_t;
typedef __attribute__((ext_vector_type(16))) __bf16 v16bf;
typedef __attribute__((ext_vector_type(4)))  __bf16 v4bf;
typedef __attribute__((ext_vector_type(8)))  float  v8f;
typedef __attribute__((ext_vector_type(4)))  float  v4f;
typedef __attribute__((ext_vector_type(4)))  int    v4i;

#define T_DIM 512
#define B_DIM 32
#define F_DIM 4096
#define H_DIM 128
#define G_DIM 512   // 4*H
#define NC 3        // NUM_CLASSES + 1

union FragU { v4i i[2]; v16bf v; };

__device__ __forceinline__ v16bf load_frag(const bf16_t* p0, const bf16_t* p1) {
  FragU f;
  f.i[0] = *(const v4i*)p0;
  f.i[1] = *(const v4i*)p1;
  return f.v;
}

__device__ __forceinline__ float fast_sigmoid(float x) {
  return __builtin_amdgcn_rcpf(1.0f + __expf(-x));
}
__device__ __forceinline__ float fast_tanh(float x) {
#if __has_builtin(__builtin_amdgcn_tanhf)
  return __builtin_amdgcn_tanhf(x);
#else
  return 1.0f - 2.0f * __builtin_amdgcn_rcpf(1.0f + __expf(2.0f * x));
#endif
}

// ---------------------------------------------------------------------------
// Kernel 0: transpose f32 [K][N] -> bf16 [N][K]
// ---------------------------------------------------------------------------
__global__ void transpose_bf16_kernel(const float* __restrict__ in,
                                      bf16_t* __restrict__ out, int K, int N) {
  int tid = blockIdx.x * blockDim.x + threadIdx.x;
  int total = K * N;
  if (tid >= total) return;
  int n = tid / K;
  int k = tid - n * K;
  out[tid] = (bf16_t)in[(size_t)k * N + n];
}

// ---------------------------------------------------------------------------
// Kernel 1: z[T][B][4H] = x @ Wk + b   (bf16 WMMA, f32 accum)
// Block = 256 thr (8 waves), tile 128x128, K-chunks of 32.
// ---------------------------------------------------------------------------
__global__ void __launch_bounds__(256) gemm_z_kernel(
    const float* __restrict__ x,       // [16384][4096]
    const bf16_t* __restrict__ wk_t,   // [512][4096]
    const float* __restrict__ bias,    // [512]
    float* __restrict__ z)             // [T][B][512]
{
  __shared__ bf16_t As[128][40];   // padded: 80B stride, conflict-free
  __shared__ bf16_t Bs[128][40];

  const int tid  = threadIdx.x;
  const int lane = tid & 31;
  const int w    = tid >> 5;
  const int wm   = w & 3;    // wave M position (0..3), 32 rows each
  const int wn   = w >> 2;   // wave N position (0..1), 64 cols each
  const int Mb   = blockIdx.x * 128;
  const int Nb   = blockIdx.y * 128;
  const int half = lane >> 4;
  const int l15  = lane & 15;

  v8f acc[2][4];
#pragma unroll
  for (int i = 0; i < 2; ++i)
#pragma unroll
    for (int j = 0; j < 4; ++j)
      acc[i][j] = (v8f){0.f,0.f,0.f,0.f,0.f,0.f,0.f,0.f};

  v4f areg[4];
  v4i breg[2];

  auto load_global = [&](int k0) {
#pragma unroll
    for (int i = 0; i < 4; ++i) {
      int idx4 = tid + i * 256;              // 1024 float4 = 128x32 f32
      int row  = idx4 >> 3;
      int col  = (idx4 & 7) << 2;
      areg[i] = *(const v4f*)(x + (size_t)(Mb + row) * F_DIM + k0 + col);
    }
#pragma unroll
    for (int i = 0; i < 2; ++i) {
      int ch  = tid + i * 256;               // 512 x 16B = 128x32 bf16
      int row = ch >> 2;
      int col = (ch & 3) << 3;
      breg[i] = *(const v4i*)(wk_t + (size_t)(Nb + row) * F_DIM + k0 + col);
    }
  };
  auto store_lds = [&]() {
#pragma unroll
    for (int i = 0; i < 4; ++i) {
      int idx4 = tid + i * 256;
      int row  = idx4 >> 3;
      int col  = (idx4 & 7) << 2;
      v4bf b4;
      b4.x = (bf16_t)areg[i].x; b4.y = (bf16_t)areg[i].y;
      b4.z = (bf16_t)areg[i].z; b4.w = (bf16_t)areg[i].w;
      *(v4bf*)(&As[row][col]) = b4;
    }
#pragma unroll
    for (int i = 0; i < 2; ++i) {
      int ch  = tid + i * 256;
      int row = ch >> 2;
      int col = (ch & 3) << 3;
      *(v4i*)(&Bs[row][col]) = breg[i];
    }
  };

  load_global(0);
  for (int kc = 0; kc < F_DIM / 32; ++kc) {
    __syncthreads();
    store_lds();
    __syncthreads();
    if (kc + 1 < F_DIM / 32) load_global((kc + 1) * 32);

    v16bf afrag[2], bfrag[4];
#pragma unroll
    for (int i = 0; i < 2; ++i) {
      int row = wm * 32 + i * 16 + l15;
      int ka  = half ? 8 : 0;                // A: K0-7/16-23 vs K8-15/24-31
      afrag[i] = load_frag(&As[row][ka], &As[row][ka + 16]);
    }
#pragma unroll
    for (int j = 0; j < 4; ++j) {
      int row = wn * 64 + j * 16 + l15;
      int kb  = half ? 8 : 0;                // B: K0-7/16-23 vs K8-15/24-31
      bfrag[j] = load_frag(&Bs[row][kb], &Bs[row][kb + 16]);
    }
#pragma unroll
    for (int i = 0; i < 2; ++i)
#pragma unroll
      for (int j = 0; j < 4; ++j)
        acc[i][j] = __builtin_amdgcn_wmma_f32_16x16x32_bf16(
            false, afrag[i], false, bfrag[j], (short)0, acc[i][j], false, false);
  }

  // epilogue: + bias, scatter rows m=b*T+t into z[T][B][4H]
#pragma unroll
  for (int i = 0; i < 2; ++i)
#pragma unroll
    for (int j = 0; j < 4; ++j) {
      int ncol = Nb + wn * 64 + j * 16 + l15;
      float bv = bias[ncol];
#pragma unroll
      for (int r = 0; r < 8; ++r) {
        int m = Mb + wm * 32 + i * 16 + r + 8 * half;
        int bidx = m >> 9;        // m / T
        int t    = m & 511;       // m % T
        z[((size_t)(t * B_DIM + bidx)) * G_DIM + ncol] = acc[i][j][r] + bv;
      }
    }
}

// ---------------------------------------------------------------------------
// Kernel 2: sequential LSTM recurrence. One workgroup, 16 waves.
// Wave (mt, jt) owns gate tiles {jt, jt+8, jt+16, jt+24} x M-tile mt,
// so i/f/c/o for its (m, j) block live in its own registers; c in regs.
// Wr (bf16, padded) + h (bf16) in dynamic LDS.
// ---------------------------------------------------------------------------
#define WR_PAD 136                                // 272B row stride
#define SMEM_WR (G_DIM * WR_PAD)                  // bf16 elements
#define SMEM_H  (B_DIM * WR_PAD)
#define SMEM_BYTES ((SMEM_WR + SMEM_H) * 2)

__global__ void __launch_bounds__(512) lstm_rec_kernel(
    const float* __restrict__ z,       // [T][32][512]
    const bf16_t* __restrict__ wr_t,   // [512][128]
    float* __restrict__ hs)            // [T][32][128]
{
  extern __shared__ char smem[];
  bf16_t* WrS = (bf16_t*)smem;                 // [512][136]
  bf16_t* hS  = (bf16_t*)smem + SMEM_WR;       // [32][136]

  const int tid  = threadIdx.x;
  const int lane = tid & 31;
  const int w    = tid >> 5;     // 0..15
  const int mt   = w >> 3;       // 0..1
  const int jt   = w & 7;        // 0..7
  const int half = lane >> 4;
  const int l15  = lane & 15;

  // Preload Wr into LDS (one row per thread, 8 x 16B)
  {
    int row = tid;
#pragma unroll
    for (int i = 0; i < 8; ++i)
      *(v4i*)(WrS + row * WR_PAD + i * 8) =
          *(const v4i*)(wr_t + (size_t)row * H_DIM + i * 8);
  }
  // h0 = 0
  for (int idx = tid; idx < SMEM_H; idx += 512) hS[idx] = (bf16_t)0.0f;
  __syncthreads();

  float c[8];
#pragma unroll
  for (int r = 0; r < 8; ++r) c[r] = 0.0f;

  const int nt0 = jt, nt1 = jt + 8, nt2 = jt + 16, nt3 = jt + 24;
  const int ntile[4] = { nt0, nt1, nt2, nt3 };
  const int hcol = jt * 16 + l15;

  for (int t = 0; t < T_DIM; ++t) {
    // issue z loads first; consumed only after the WMMA chain
    float zv[4][8];
    const float* zt = z + (size_t)t * B_DIM * G_DIM;
#pragma unroll
    for (int g = 0; g < 4; ++g) {
      int ncol = ntile[g] * 16 + l15;
#pragma unroll
      for (int r = 0; r < 8; ++r) {
        int m = mt * 16 + r + 8 * half;
        zv[g][r] = zt[m * G_DIM + ncol];
      }
    }

    // A fragments from h (bf16 in LDS)
    v16bf af[4];
#pragma unroll
    for (int kc = 0; kc < 4; ++kc) {
      int row = mt * 16 + l15;
      int ka  = kc * 32 + (half ? 8 : 0);
      af[kc] = load_frag(hS + row * WR_PAD + ka, hS + row * WR_PAD + ka + 16);
    }

    v8f acc[4];
#pragma unroll
    for (int g = 0; g < 4; ++g)
      acc[g] = (v8f){0.f,0.f,0.f,0.f,0.f,0.f,0.f,0.f};
#pragma unroll
    for (int g = 0; g < 4; ++g) {
      int brow = ntile[g] * 16 + l15;
#pragma unroll
      for (int kc = 0; kc < 4; ++kc) {
        int kb = kc * 32 + (half ? 8 : 0);
        v16bf bf = load_frag(WrS + brow * WR_PAD + kb,
                             WrS + brow * WR_PAD + kb + 16);
        acc[g] = __builtin_amdgcn_wmma_f32_16x16x32_bf16(
            false, af[kc], false, bf, (short)0, acc[g], false, false);
      }
    }

    // gate math + state update, all in registers
    float hv[8];
#pragma unroll
    for (int r = 0; r < 8; ++r) {
      float gi = acc[0][r] + zv[0][r];
      float gf = acc[1][r] + zv[1][r];
      float gc = acc[2][r] + zv[2][r];
      float go = acc[3][r] + zv[3][r];
      float i_ = fast_sigmoid(gi);
      float f_ = fast_sigmoid(gf);
      float cc = fast_tanh(gc);
      float o_ = fast_sigmoid(go);
      c[r]  = f_ * c[r] + i_ * cc;
      hv[r] = o_ * fast_tanh(c[r]);
    }

    __syncthreads();   // everyone done reading hS(t)
    float* hst = hs + (size_t)t * B_DIM * H_DIM;
#pragma unroll
    for (int r = 0; r < 8; ++r) {
      int m = mt * 16 + r + 8 * half;
      hS[m * WR_PAD + hcol] = (bf16_t)hv[r];
      hst[m * H_DIM + hcol] = hv[r];
    }
    __syncthreads();   // hS(t+1) visible
  }
}

// ---------------------------------------------------------------------------
// Kernel 3: out[b][t][k] = hs[t][b][:] . Wd[:,k] + bd[k]
// ---------------------------------------------------------------------------
__global__ void dense_kernel(const float* __restrict__ hs,
                             const float* __restrict__ wd,
                             const float* __restrict__ bd,
                             float* __restrict__ out) {
  int tid = blockIdx.x * blockDim.x + threadIdx.x;
  if (tid >= B_DIM * T_DIM * NC) return;
  int k    = tid % NC;
  int mrow = tid / NC;            // b*T + t
  int b    = mrow >> 9;
  int t    = mrow & 511;
  const float* hrow = hs + ((size_t)t * B_DIM + b) * H_DIM;
  float s = bd[k];
#pragma unroll 8
  for (int j = 0; j < H_DIM; ++j) s += hrow[j] * wd[j * NC + k];
  out[tid] = s;
}

// ---------------------------------------------------------------------------
extern "C" void kernel_launch(void* const* d_in, const int* in_sizes, int n_in,
                              void* d_out, int out_size, void* d_ws, size_t ws_size,
                              hipStream_t stream) {
  const float* x    = (const float*)d_in[0];
  const float* Wk   = (const float*)d_in[1];
  const float* Wr   = (const float*)d_in[2];
  const float* bias = (const float*)d_in[3];
  const float* Wd   = (const float*)d_in[4];
  const float* bd   = (const float*)d_in[5];
  float* out = (float*)d_out;

  char* ws = (char*)d_ws;
  const size_t Z_BYTES   = (size_t)T_DIM * B_DIM * G_DIM * 4;   // 32 MB
  const size_t WKT_BYTES = (size_t)G_DIM * F_DIM * 2;           // 4 MB
  const size_t WRT_BYTES = (size_t)G_DIM * H_DIM * 2;           // 128 KB
  float*  z    = (float*)ws;
  bf16_t* wk_t = (bf16_t*)(ws + Z_BYTES);
  bf16_t* wr_t = (bf16_t*)(ws + Z_BYTES + WKT_BYTES);
  float*  hs   = (float*)(ws + Z_BYTES + WKT_BYTES + WRT_BYTES);

  // Wk [4096][512] -> bf16 [512][4096]; Wr [128][512] -> bf16 [512][128]
  transpose_bf16_kernel<<<(F_DIM * G_DIM + 255) / 256, 256, 0, stream>>>(
      Wk, wk_t, F_DIM, G_DIM);
  transpose_bf16_kernel<<<(H_DIM * G_DIM + 255) / 256, 256, 0, stream>>>(
      Wr, wr_t, H_DIM, G_DIM);

  // z = x @ Wk + b  (grid: 128 M-tiles x 4 N-tiles of 128x128)
  gemm_z_kernel<<<dim3(128, 4), 256, 0, stream>>>(x, wk_t, bias, z);

  // sequential recurrence: single persistent workgroup, 148KB dynamic LDS
  hipFuncSetAttribute((const void*)lstm_rec_kernel,
                      hipFuncAttributeMaxDynamicSharedMemorySize,
                      (int)SMEM_BYTES);
  lstm_rec_kernel<<<1, 512, SMEM_BYTES, stream>>>(z, wr_t, hs);

  // dense head
  dense_kernel<<<(B_DIM * T_DIM * NC + 255) / 256, 256, 0, stream>>>(
      hs, Wd, bd, out);
}
